// groupLista_46059229282427
// MI455X (gfx1250) — matrix operand order repaired and checked by reading, not compiled, not run on hardware
//
#include <hip/hip_runtime.h>
#include <hip/hip_bf16.h>

// ---------------------------------------------------------------------------
// groupLista on gfx1250: everything reduced to bf16 WMMA GEMMs + small glue.
// Shapes: B=2 batches, NCH=243 (pad KP=256), P=256 filters, N=2304 patches.
// A-tile staging via Tensor Data Mover (tensor_load_to_lds) when available.
// ---------------------------------------------------------------------------

typedef __bf16 bf16;
typedef __attribute__((ext_vector_type(16))) __bf16 v16bf;
typedef __attribute__((ext_vector_type(8)))  __bf16 v8bf;
typedef __attribute__((ext_vector_type(8)))  float  v8f;
typedef __attribute__((ext_vector_type(4)))  unsigned int v4u;
typedef __attribute__((ext_vector_type(4)))  int v4i;
typedef __attribute__((ext_vector_type(8)))  int v8i;

#define BATCH 2
#define CH    3
#define HIMG  56
#define KSZ   9
#define HO    48
#define NPAT  (HO*HO)        // 2304
#define NCH   243
#define KP    256
#define PF    256
#define UNFOLD 12
#define FREQ   6

#define BM 64
#define BN 128
#define BK 32
#define LDSPAD 8             // halves of pad per row (== 4 DWORDs, matches TDM pad)

#if defined(__gfx1250__) && __has_builtin(__builtin_amdgcn_tensor_load_to_lds)
#define HAVE_TDM 1
#else
#define HAVE_TDM 0
#endif

// epilogue selectors
#define EPI_F32  0
#define EPI_BF16 1
#define EPI_Z    2
#define EPI_PROX 3
#define EPI_GRAM 4
#define EPI_OUT  5
#define EPI_YT   6

struct GP {
    const bf16* A; const bf16* B;
    long sA, sB;               // per-batch element strides for A/B (0 = shared weight)
    int  lda, ldb, ldc;
    float* f0; bf16* b0;       // primary outputs (canonical row*ldc+col + bz*sC)
    long  sC;
    float* f1;                 // extra f32 out (Ysq atomics for EPI_YT)
    const float* aux0;         // Z: gamma_f32 | PROX: Z_f32
    const float* aux1;         // Z: A0
    const float* theta;        // PROX: lambda row
    const float* sqv;          // GRAM: per-patch sq-norms
    const float* meanv;        // OUT: per-patch mean
    const float* std0;         // YT: per-channel scale
    long sYt;                  // YT: per-batch stride of Yt
    long sN;                   // per-batch stride for sq/mean/Ysq (=NPAT)
};

template<int EPI, bool TRANSB>
__global__ __launch_bounds__(256) void gemm_k(GP p, int Ktiles)
{
    __shared__ bf16 As [BM][BK + LDSPAD];   // k-major A tile (row stride 80 B)
    __shared__ bf16 BsT[BN][BK + LDSPAD];   // n-major B tile -> contiguous frag reads

    const int bz = blockIdx.z;
    const int m0 = blockIdx.y * BM;
    const int n0 = blockIdx.x * BN;

    const int t    = threadIdx.x;
    const int lane = t & 31;
    const int wid  = t >> 5;
    const int wm   = (wid >> 2) << 5;      // 0 / 32
    const int wn   = (wid & 3) << 5;       // 0..96
    const int lm   = lane & 15;
    const bool hi  = (lane & 16) != 0;
    const int hiA  = hi ? 8 : 0;           // A frag K-base within 0..15 half-group
    const int hiB  = hi ? 16 : 0;          // unused (B frag reads use hiA, as A does)

    const bf16* Ag = p.A + (long)bz * p.sA;
    const bf16* Bg = p.B + (long)bz * p.sB;

    v8f acc[2][2] = {};

    const int ar = t >> 2, ak = (t & 3) << 3;        // A fill (fallback): 64 rows x 4 groups of 8
    const int br = t >> 3, bc = (t & 7) << 4;        // B fill (KxN src): 32 k-rows x 8 groups of 16
    const int bn = t >> 1, bk = (t & 1) << 4;        // B fill (NxK src, gram)

    for (int kt = 0; kt < Ktiles; ++kt) {
        const int k0 = kt * BK;
        __syncthreads();

        // ---- stage A tile (BM x BK) ----
#if HAVE_TDM
        if (t < 32) {
            // Tensor DMA descriptor (ISA 08_async_tensor §8): 2D tile, 2-byte
            // elements, LDS pad 4 DWORDs after every 16 DWORDs (= one BK row).
            const unsigned long long ga =
                (unsigned long long)(const void*)(Ag + (size_t)m0 * p.lda + (size_t)k0);
            const unsigned lds = (unsigned)(unsigned long long)(void*)&As[0][0];
            v4u g0;
            g0[0] = (unsigned)__builtin_amdgcn_readfirstlane(1);               // count=1
            g0[1] = (unsigned)__builtin_amdgcn_readfirstlane((int)lds);        // lds_addr
            g0[2] = (unsigned)__builtin_amdgcn_readfirstlane((int)(ga & 0xffffffffu));
            g0[3] = (unsigned)__builtin_amdgcn_readfirstlane(
                        (int)(((unsigned)(ga >> 32) & 0x1ffffffu) | (2u << 30))); // type=2
            v8i g1;
            g1[0] = __builtin_amdgcn_readfirstlane(
                        (int)((1u << 16) | (1u << 20) | (3u << 22) | (3u << 25)));
            //        data_size=2B | pad_en | pad_interval=16dw | pad_amount=4dw
            g1[1] = __builtin_amdgcn_readfirstlane((int)((unsigned)BK << 16)); // tensor_dim0=32
            g1[2] = __builtin_amdgcn_readfirstlane((int)((unsigned)BM << 16)); // tensor_dim1=64
            g1[3] = __builtin_amdgcn_readfirstlane((int)((unsigned)BK << 16)); // tile_dim0=32
            g1[4] = __builtin_amdgcn_readfirstlane((int)BM);                   // tile_dim1=64
            g1[5] = __builtin_amdgcn_readfirstlane((int)p.lda);                // dim0_stride lo32
            g1[6] = 0;
            g1[7] = 0;
            const v4i z4 = {};
#if __clang_major__ >= 23
            const v8i z8 = {};
            __builtin_amdgcn_tensor_load_to_lds(g0, g1, z4, z4, z8, 0);
#else
            __builtin_amdgcn_tensor_load_to_lds(g0, g1, z4, z4, 0);
#endif
            __builtin_amdgcn_s_wait_tensorcnt(0);
        }
#else
        {
            const bf16* src = Ag + (size_t)(m0 + ar) * p.lda + k0 + ak;
            *(v8bf*)&As[ar][ak] = *(const v8bf*)src;
            __builtin_prefetch(src + BK, 0, 1);
        }
#endif

        // ---- stage B tile into n-major LDS (BN x BK) ----
        if (!TRANSB) {
            const bf16* src = Bg + (size_t)(k0 + br) * p.ldb + n0 + bc;
            v8bf x0 = *(const v8bf*)src;
            v8bf x1 = *(const v8bf*)(src + 8);
#pragma unroll
            for (int i = 0; i < 8; ++i) {
                BsT[bc + i][br]     = x0[i];
                BsT[bc + 8 + i][br] = x1[i];
            }
            __builtin_prefetch(src + (size_t)BK * p.ldb, 0, 1);
        } else {
            // B[k][n] = Bt[n][k]: contiguous along k both in global and LDS
            const bf16* src = Bg + (size_t)(n0 + bn) * p.ldb + k0 + bk;
            *(v8bf*)&BsT[bn][bk]     = *(const v8bf*)src;
            *(v8bf*)&BsT[bn][bk + 8] = *(const v8bf*)(src + 8);
            __builtin_prefetch(src + BK, 0, 1);
        }
        __syncthreads();

        // ---- fragments per ISA 7.12.2 VGPR layouts (vectorized LDS reads) ----
        v16bf afr[2], bfr[2];
#pragma unroll
        for (int mt = 0; mt < 2; ++mt) {
            const int row = wm + mt * 16 + lm;
            v8bf lo = *(const v8bf*)&As[row][hiA];        // K = hiA+0..7
            v8bf hi8 = *(const v8bf*)&As[row][16 + hiA];  // K = 16+hiA+0..7
            afr[mt] = __builtin_shufflevector(lo, hi8,
                0, 1, 2, 3, 4, 5, 6, 7, 8, 9, 10, 11, 12, 13, 14, 15);
        }
#pragma unroll
        for (int nt = 0; nt < 2; ++nt) {
            const int col = wn + nt * 16 + lm;
            v8bf lo = *(const v8bf*)&BsT[col][hiA];       // K = hiA+0..7
            v8bf hi8 = *(const v8bf*)&BsT[col][16 + hiA]; // K = 16+hiA+0..7
            bfr[nt] = __builtin_shufflevector(lo, hi8,
                0, 1, 2, 3, 4, 5, 6, 7, 8, 9, 10, 11, 12, 13, 14, 15);
        }
#pragma unroll
        for (int mt = 0; mt < 2; ++mt)
#pragma unroll
            for (int nt = 0; nt < 2; ++nt)
                acc[mt][nt] = __builtin_amdgcn_wmma_f32_16x16x32_bf16(
                    false, afr[mt], false, bfr[nt], (short)0, acc[mt][nt], false, false);
    }

    // ---- epilogue (D layout: row = base + 8*hi + r, col = base + lane&15) ----
    const long cb = (long)bz * p.sC;
#pragma unroll
    for (int mt = 0; mt < 2; ++mt)
#pragma unroll
        for (int nt = 0; nt < 2; ++nt)
#pragma unroll
            for (int r = 0; r < 8; ++r) {
                const int row = m0 + wm + mt * 16 + (hi ? 8 : 0) + r;
                const int col = n0 + wn + nt * 16 + lm;
                const float c = acc[mt][nt][r];
                const long idx = cb + (long)row * p.ldc + col;
                if (EPI == EPI_F32) {
                    p.f0[idx] = c;
                } else if (EPI == EPI_BF16) {
                    p.b0[idx] = (bf16)c;
                } else if (EPI == EPI_Z) {
                    const float z = p.aux0[idx] + p.aux1[idx] - c;  // gamma + A0 - G*gamma
                    p.f0[idx] = z;
                    p.b0[idx] = (bf16)(z * z);
                } else if (EPI == EPI_PROX) {
                    // support == 1 (softmax cols sum to 1); factor = relu(1 - th/sqrt(n2+eps))
                    const float z = p.aux0[idx];
                    float fct = 1.0f - p.theta[row] / sqrtf(c + 1e-8f);
                    fct = fmaxf(fct, 0.0f);
                    const float g = fct * z;
                    p.f0[idx] = g;
                    p.b0[idx] = (bf16)g;
                } else if (EPI == EPI_GRAM) {
                    // logit = -(sq_i + sq_j - 2 g_ij)
                    const long bN = (long)bz * p.sN;
                    p.f0[idx] = 2.0f * c - p.sqv[bN + row] - p.sqv[bN + col];
                } else if (EPI == EPI_OUT) {
                    p.f0[idx] = c + p.meanv[(long)bz * p.sN + col];
                } else if (EPI == EPI_YT) {
                    const float val = (row < NCH) ? c * p.std0[row] : 0.0f;
                    p.b0[(long)bz * p.sYt + ((long)col << 8) + row] = (bf16)val;
                    if (row < NCH)
                        atomicAdd(&p.f1[(long)bz * p.sN + col], val * val);
                }
            }
}

// --- pack weights into padded bf16 (zeros in pad rows/cols kill the K=256 loop) ---
__global__ void pack_w_k(const float* Wa, const float* Wd, const float* Ww,
                         bf16* WaB, bf16* WdB, bf16* WwB)
{
    const int t = blockIdx.x * blockDim.x + threadIdx.x;
    if (t >= 256 * 256) return;
    const int r = t >> 8, c = t & 255;
    WaB[t] = (bf16)((c < NCH) ? Wa[r * NCH + c] : 0.0f);      // (P x KP)
    WdB[t] = (bf16)((r < NCH) ? Wd[r * 256 + c] : 0.0f);      // (KP x P)
    WwB[t] = (bf16)((r < NCH) ? Ww[r * 256 + c] : 0.0f);      // (KP(chan) x P)
}

// --- im2col + mean-subtract; emit Icol (KPxN, k-major, bf16), Xt (NxKP scaled), sq, mean ---
__global__ void im2col_k(const float* I, const float* std0,
                         bf16* IcolB, bf16* Xt, float* sq, float* meanv)
{
    const int t = blockIdx.x * blockDim.x + threadIdx.x;
    if (t >= BATCH * NPAT) return;
    const int b = t / NPAT, j = t % NPAT;
    const int oy = j / HO, ox = j % HO;
    const float* Ib = I + (size_t)b * CH * HIMG * HIMG;

    float s = 0.0f;
    for (int c = 0; c < CH; ++c)
        for (int ki = 0; ki < KSZ; ++ki)
            for (int kj = 0; kj < KSZ; ++kj)
                s += Ib[c * HIMG * HIMG + (oy + ki) * HIMG + (ox + kj)];
    const float mean = s * (1.0f / (float)NCH);
    meanv[t] = mean;

    bf16* xrow = Xt + (size_t)t * KP;
    float sqs = 0.0f;
    for (int c = 0; c < CH; ++c)
        for (int ki = 0; ki < KSZ; ++ki)
            for (int kj = 0; kj < KSZ; ++kj) {
                const int kidx = c * 81 + ki * KSZ + kj;
                const float val = Ib[c * HIMG * HIMG + (oy + ki) * HIMG + (ox + kj)] - mean;
                IcolB[((size_t)b * KP + kidx) * NPAT + j] = (bf16)val;
                const float xs = val * std0[kidx];
                xrow[kidx] = (bf16)xs;
                sqs += xs * xs;
            }
    for (int kidx = NCH; kidx < KP; ++kidx) {
        IcolB[((size_t)b * KP + kidx) * NPAT + j] = (bf16)0.0f;
        xrow[kidx] = (bf16)0.0f;
    }
    sq[t] = sqs;
}

// --- column softmax over logits (axis i), optional sigmoid(nu) mix into sim_bf16 ---
__global__ void softmax_mix_k(const float* lgt, bf16* simb, const float* nu_p, int use_mix)
{
    const int t = blockIdx.x * blockDim.x + threadIdx.x;
    if (t >= BATCH * NPAT) return;
    const int b = t / NPAT, j = t % NPAT;
    const float* L = lgt + (size_t)b * NPAT * NPAT + j;
    bf16* S = simb + (size_t)b * NPAT * NPAT + j;

    float m = -1e30f;
    for (int i = 0; i < NPAT; ++i) m = fmaxf(m, L[(size_t)i * NPAT]);
    float s = 0.0f;
    for (int i = 0; i < NPAT; ++i) s += __expf(L[(size_t)i * NPAT] - m);
    const float inv = 1.0f / s;

    float tm = 1.0f;
    if (use_mix) tm = 1.0f / (1.0f + __expf(-nu_p[0]));
    for (int i = 0; i < NPAT; ++i) {
        const float v = __expf(L[(size_t)i * NPAT] - m) * inv;
        const float old = use_mix ? (float)S[(size_t)i * NPAT] : 0.0f;
        S[(size_t)i * NPAT] = (bf16)((1.0f - tm) * old + tm * v);
    }
}

__global__ void z_init_k(const float* A0, float* Zf, bf16* Z2b, long n)
{
    const long i = (long)blockIdx.x * blockDim.x + threadIdx.x;
    if (i >= n) return;
    const float z = A0[i];
    Zf[i] = z;
    Z2b[i] = (bf16)(z * z);
}

__global__ void zero_f32_k(float* p, long n)
{
    const long i = (long)blockIdx.x * blockDim.x + threadIdx.x;
    if (i < n) p[i] = 0.0f;
}

// --- fold with overlap averaging ---
__global__ void col2im_k(const float* outc, float* out)
{
    const int t = blockIdx.x * blockDim.x + threadIdx.x;
    if (t >= BATCH * CH * HIMG * HIMG) return;
    const int b = t / (CH * HIMG * HIMG);
    const int rem = t % (CH * HIMG * HIMG);
    const int c = rem / (HIMG * HIMG);
    const int y = (rem % (HIMG * HIMG)) / HIMG;
    const int x = rem % HIMG;
    float acc = 0.0f, cnt = 0.0f;
    for (int ki = 0; ki < KSZ; ++ki) {
        const int oy = y - ki;
        if (oy < 0 || oy >= HO) continue;
        for (int kj = 0; kj < KSZ; ++kj) {
            const int ox = x - kj;
            if (ox < 0 || ox >= HO) continue;
            const int kidx = c * 81 + ki * KSZ + kj;
            acc += outc[((size_t)b * KP + kidx) * NPAT + oy * HO + ox];
            cnt += 1.0f;
        }
    }
    out[t] = acc / cnt;
}

extern "C" void kernel_launch(void* const* d_in, const int* in_sizes, int n_in,
                              void* d_out, int out_size, void* d_ws, size_t ws_size,
                              hipStream_t stream)
{
    (void)in_sizes; (void)n_in; (void)out_size; (void)ws_size;
    const float* I    = (const float*)d_in[0];
    const float* Wa   = (const float*)d_in[1];
    const float* Wd   = (const float*)d_in[2];
    const float* Ww   = (const float*)d_in[3];
    const float* std0 = (const float*)d_in[4];
    const float* lmb  = (const float*)d_in[5];
    const float* nu   = (const float*)d_in[6];
    float* out = (float*)d_out;

    // ---- workspace carve ----
    char* w = (char*)d_ws;
    auto take = [&](size_t bytes) -> void* {
        void* r = (void*)w;
        w += (bytes + 255) & ~(size_t)255;
        return r;
    };
    const long PN = (long)PF * NPAT;          // 256*2304
    const long NN = (long)NPAT * NPAT;        // 2304*2304
    bf16*  IcolB = (bf16*) take((size_t)BATCH * KP * NPAT * 2);
    bf16*  Xt    = (bf16*) take((size_t)BATCH * NPAT * KP * 2);
    float* sq    = (float*)take((size_t)BATCH * NPAT * 4);
    float* meanv = (float*)take((size_t)BATCH * NPAT * 4);
    bf16*  WaB   = (bf16*) take(256 * 256 * 2);
    bf16*  WdB   = (bf16*) take(256 * 256 * 2);
    bf16*  WwB   = (bf16*) take(256 * 256 * 2);
    bf16*  Gb    = (bf16*) take(256 * 256 * 2);
    float* A0    = (float*)take((size_t)BATCH * PN * 4);
    float* Zf    = (float*)take((size_t)BATCH * PN * 4);
    bf16*  Z2b   = (bf16*) take((size_t)BATCH * PN * 2);
    float* gf    = (float*)take((size_t)BATCH * PN * 4);
    bf16*  gb    = (bf16*) take((size_t)BATCH * PN * 2);
    float* lgt   = (float*)take((size_t)BATCH * NN * 4);
    bf16*  simb  = (bf16*) take((size_t)BATCH * NN * 2);
    bf16*  Yt    = (bf16*) take((size_t)BATCH * NPAT * KP * 2);
    float* Ysq   = (float*)take((size_t)BATCH * NPAT * 4);
    float* outc  = (float*)take((size_t)BATCH * KP * NPAT * 4);

    // ---- setup ----
    pack_w_k<<<(256 * 256 + 255) / 256, 256, 0, stream>>>(Wa, Wd, Ww, WaB, WdB, WwB);
    im2col_k<<<(BATCH * NPAT + 127) / 128, 128, 0, stream>>>(I, std0, IcolB, Xt, sq, meanv);

    GP p{};
    // G = Wa @ Wd  (256x256x256) -> bf16
    p = GP{}; p.A = WaB; p.B = WdB; p.sA = 0; p.sB = 0;
    p.lda = 256; p.ldb = 256; p.ldc = 256; p.b0 = Gb; p.sC = 0;
    gemm_k<EPI_BF16, false><<<dim3(256 / BN, 256 / BM, 1), 256, 0, stream>>>(p, KP / BK);

    // A0 = Wa @ Icol  (256 x 2304, K=256)
    p = GP{}; p.A = WaB; p.B = IcolB; p.sA = 0; p.sB = (long)KP * NPAT;
    p.lda = 256; p.ldb = NPAT; p.ldc = NPAT; p.f0 = A0; p.sC = PN;
    gemm_k<EPI_F32, false><<<dim3(NPAT / BN, PF / BM, BATCH), 256, 0, stream>>>(p, KP / BK);

    // logits = 2*Xt.Xt^T - sq_i - sq_j  (2304x2304, K=256)
    p = GP{}; p.A = Xt; p.B = Xt; p.sA = (long)NPAT * KP; p.sB = (long)NPAT * KP;
    p.lda = KP; p.ldb = KP; p.ldc = NPAT; p.f0 = lgt; p.sC = NN; p.sqv = sq; p.sN = NPAT;
    gemm_k<EPI_GRAM, true><<<dim3(NPAT / BN, NPAT / BM, BATCH), 256, 0, stream>>>(p, KP / BK);

    softmax_mix_k<<<(BATCH * NPAT + 255) / 256, 256, 0, stream>>>(lgt, simb, nu, 0);

    // gamma_0 = prox(A0, lmbdas[0], sim)
    z_init_k<<<(unsigned)((BATCH * PN + 255) / 256), 256, 0, stream>>>(A0, Zf, Z2b, BATCH * PN);
    p = GP{}; p.A = Z2b; p.B = simb; p.sA = PN; p.sB = NN;
    p.lda = NPAT; p.ldb = NPAT; p.ldc = NPAT; p.f0 = gf; p.b0 = gb; p.sC = PN;
    p.aux0 = Zf; p.theta = lmb;
    gemm_k<EPI_PROX, false><<<dim3(NPAT / BN, PF / BM, BATCH), 256, 0, stream>>>(p, NPAT / BK);

    for (int k = 0; k < UNFOLD - 1; ++k) {
        if (k % FREQ == 0 && k != 0) {
            // refresh: Yt = (Ww @ gamma * std0)^T, Ysq; gram -> logits; softmax-mix
            zero_f32_k<<<(BATCH * NPAT + 255) / 256, 256, 0, stream>>>(Ysq, BATCH * NPAT);
            p = GP{}; p.A = WwB; p.B = gb; p.sA = 0; p.sB = PN;
            p.lda = 256; p.ldb = NPAT; p.ldc = NPAT; p.sC = PN;
            p.b0 = Yt; p.sYt = (long)NPAT * KP; p.f1 = Ysq; p.std0 = std0; p.sN = NPAT;
            gemm_k<EPI_YT, false><<<dim3(NPAT / BN, KP / BM, BATCH), 256, 0, stream>>>(p, KP / BK);

            p = GP{}; p.A = Yt; p.B = Yt; p.sA = (long)NPAT * KP; p.sB = (long)NPAT * KP;
            p.lda = KP; p.ldb = KP; p.ldc = NPAT; p.f0 = lgt; p.sC = NN; p.sqv = Ysq; p.sN = NPAT;
            gemm_k<EPI_GRAM, true><<<dim3(NPAT / BN, NPAT / BM, BATCH), 256, 0, stream>>>(p, KP / BK);

            softmax_mix_k<<<(BATCH * NPAT + 255) / 256, 256, 0, stream>>>(lgt, simb, nu, 1);
        }
        // Z = gamma + A0 - G @ gamma
        p = GP{}; p.A = Gb; p.B = gb; p.sA = 0; p.sB = PN;
        p.lda = 256; p.ldb = NPAT; p.ldc = NPAT; p.f0 = Zf; p.b0 = Z2b; p.sC = PN;
        p.aux0 = gf; p.aux1 = A0;
        gemm_k<EPI_Z, false><<<dim3(NPAT / BN, PF / BM, BATCH), 256, 0, stream>>>(p, KP / BK);

        // gamma = prox(Z, lmbdas[k+1], sim)
        p = GP{}; p.A = Z2b; p.B = simb; p.sA = PN; p.sB = NN;
        p.lda = NPAT; p.ldb = NPAT; p.ldc = NPAT; p.f0 = gf; p.b0 = gb; p.sC = PN;
        p.aux0 = Zf; p.theta = lmb + (size_t)(k + 1) * PF;
        gemm_k<EPI_PROX, false><<<dim3(NPAT / BN, PF / BM, BATCH), 256, 0, stream>>>(p, NPAT / BK);
    }

    // out_cols = Ww @ gamma + mean
    p = GP{}; p.A = WwB; p.B = gb; p.sA = 0; p.sB = PN;
    p.lda = 256; p.ldb = NPAT; p.ldc = NPAT; p.f0 = outc; p.sC = (long)KP * NPAT;
    p.meanv = meanv; p.sN = NPAT;
    gemm_k<EPI_OUT, false><<<dim3(NPAT / BN, KP / BM, BATCH), 256, 0, stream>>>(p, KP / BK);

    col2im_k<<<(BATCH * CH * HIMG * HIMG + 255) / 256, 256, 0, stream>>>(outc, out);
}
